// SemanticCodebook_88656714924738
// MI455X (gfx1250) — compile-verified, matrix operand order checked by reading, not run on hardware
//
#include <hip/hip_runtime.h>

// ---------------------------------------------------------------------------
// VQ codebook assignment for MI455X (gfx1250, wave32, WMMA).
// codes = argmin_K (|e|^2 - 2 x.e)   [x^2 dropped: constant per row]
// quantized = emb[codes].
// GEMM via split-bf16 (hi+lo) v_wmma_f32_16x16x32_bf16, fp32 accumulate.
// Each 4-wave WG stages 64 x-rows in LDS; waves split the 8192 codes 4-way
// and each wave multiplies its B-fragment against all four 16-row A tiles,
// so the 16MB bf16 codebook is read exactly once per WG (4GB total from L2).
// ---------------------------------------------------------------------------

#define VQ_EPS 1e-5f

typedef __attribute__((ext_vector_type(16))) __bf16 v16bf;
typedef __attribute__((ext_vector_type(8)))  __bf16 v8bf;
typedef __attribute__((ext_vector_type(8)))  float  v8f;

static constexpr int Bk    = 16;
static constexpr int Dk    = 512;
static constexpr int Tk    = 1024;
static constexpr int Nrows = Bk * Tk;   // 16384
static constexpr int Kc    = 8192;

// workspace layout (bytes)
static constexpr size_t OFF_HI    = 0;
static constexpr size_t OFF_LO    = OFF_HI + (size_t)Kc * Dk * 2;      // +8MB
static constexpr size_t OFF_E2    = OFF_LO + (size_t)Kc * Dk * 2;      // +8MB
static constexpr size_t OFF_INV   = OFF_E2 + (size_t)Kc * 4;
static constexpr size_t OFF_CODES = OFF_INV + (size_t)Kc * 4;

// dynamic LDS: 64x512 bf16 hi + lo (128KB) + cross-wave argmin scratch (2KB)
static constexpr int kLdsXBytes  = 64 * Dk * 2 * 2;            // 131072
static constexpr int kLdsRedOff  = kLdsXBytes;
static constexpr int kLdsBytes   = kLdsXBytes + 64 * 4 * 8;    // 133120

__device__ __forceinline__ __bf16 f32_to_bf16(float f) {
  unsigned u = __builtin_bit_cast(unsigned, f);
  unsigned r = (u + 0x7FFFu + ((u >> 16) & 1u)) >> 16;   // round-nearest-even
  unsigned short s = (unsigned short)r;
  return __builtin_bit_cast(__bf16, s);
}
__device__ __forceinline__ float bf16_to_f32(__bf16 h) {
  unsigned short s = __builtin_bit_cast(unsigned short, h);
  unsigned u = ((unsigned)s) << 16;
  return __builtin_bit_cast(float, u);
}

// Assemble a 16-bit WMMA fragment lane-slice from two contiguous 16B runs.
__device__ __forceinline__ v16bf frag16(const __bf16* p0, const __bf16* p1) {
  v8bf a = *(const v8bf*)p0;
  v8bf b = *(const v8bf*)p1;
  v16bf r;
#pragma unroll
  for (int i = 0; i < 8; ++i) { r[i] = a[i]; r[i + 8] = b[i]; }
  return r;
}

// --- kernel 1: codebook prep: emb = es/clip(usage), bf16 hi/lo, e2, 1/usage -
__global__ __launch_bounds__(128)
void vq_prep_codebook(const float* __restrict__ es,
                      const float* __restrict__ usage,
                      __bf16* __restrict__ hi, __bf16* __restrict__ lo,
                      float* __restrict__ e2, float* __restrict__ inv) {
  __shared__ float red[128];
  int k = blockIdx.x;
  float u = usage[k];
  u = u < VQ_EPS ? VQ_EPS : u;
  float iu = 1.0f / u;
  float acc = 0.f;
#pragma unroll
  for (int j = 0; j < 4; ++j) {
    int d = threadIdx.x + 128 * j;
    float e = es[(size_t)k * Dk + d] * iu;
    __bf16 h = f32_to_bf16(e);
    hi[(size_t)k * Dk + d] = h;
    lo[(size_t)k * Dk + d] = f32_to_bf16(e - bf16_to_f32(h));
    acc += e * e;
  }
  red[threadIdx.x] = acc;
  __syncthreads();
  for (int s = 64; s > 0; s >>= 1) {
    if ((int)threadIdx.x < s) red[threadIdx.x] += red[threadIdx.x + s];
    __syncthreads();
  }
  if (threadIdx.x == 0) { e2[k] = red[0]; inv[k] = iu; }
}

// --- kernel 2: fused GEMM + argmin -----------------------------------------
__global__ __launch_bounds__(128)
void vq_argmin(const float* __restrict__ x,
               const __bf16* __restrict__ ehi, const __bf16* __restrict__ elo,
               const float* __restrict__ e2,
               int* __restrict__ codes_i, float* __restrict__ codes_f) {
  extern __shared__ __align__(16) char smem[];
  __bf16* ldsHi = (__bf16*)smem;            // 64 rows x 512
  __bf16* ldsLo = ldsHi + 64 * Dk;          // 64 rows x 512
  float*  redD  = (float*)(smem + kLdsRedOff);        // [64 rows][4 waves]
  int*    redI  = (int*)(smem + kLdsRedOff + 64 * 16);// [64 rows][4 waves]

  const int rowBase = blockIdx.x * 64;      // 64-row tile never crosses b
  const int b = rowBase >> 10;
  const int tBase = rowBase & 1023;

  { // stage + convert + hi/lo split (coalesced over t == m)
    int m = threadIdx.x & 63;
    int dphase = threadIdx.x >> 6;          // 0..1
    const float* xp = x + (size_t)b * Dk * Tk + tBase + m;
    for (int d = dphase; d < Dk; d += 2) {
      float v = xp[(size_t)d * Tk];
      __bf16 h = f32_to_bf16(v);
      ldsHi[m * Dk + d] = h;
      ldsLo[m * Dk + d] = f32_to_bf16(v - bf16_to_f32(h));
    }
  }
  __syncthreads();

  const int wave = threadIdx.x >> 5;
  const int lane = threadIdx.x & 31;
  const int lrow = lane & 15;
  const int half = lane >> 4;
  const int hofs = half * 8;                // ISA 16-bit A/B lane K-offset

  // wave w owns codes [w*2048, (w+1)*2048): disjoint + ordered -> tie-break ok
  const int ctBeg = wave * (Kc / 16 / 4);
  const int ctEnd = ctBeg + (Kc / 16 / 4);

  float best[4][8]; int bidx[4][8];
#pragma unroll
  for (int tmi = 0; tmi < 4; ++tmi)
#pragma unroll
    for (int r = 0; r < 8; ++r) { best[tmi][r] = 3.4e38f; bidx[tmi][r] = 0; }

  for (int ct = ctBeg; ct < ctEnd; ++ct) {
    const int n = ct * 16 + lrow;                   // this lane's code column
    const __bf16* bHi = ehi + (size_t)n * Dk;
    const __bf16* bLo = elo + (size_t)n * Dk;
    __builtin_prefetch(bHi + Dk * 16, 0, 0);        // next column tile
    __builtin_prefetch(bLo + Dk * 16, 0, 0);

    v8f c0 = {}, c1 = {}, c2 = {}, c3 = {};
#pragma unroll 2
    for (int dk = 0; dk < Dk; dk += 32) {
      const int o0 = dk + hofs;            // lane K = dk + {0..7}/{8..15}
      const int o1 = dk + 16 + hofs;       // lane K = dk + 16 + {0..7}/{8..15}
      v16bf bh = frag16(bHi + o0, bHi + o1);
      v16bf bl = frag16(bLo + o0, bLo + o1);
#pragma unroll
      for (int tmi = 0; tmi < 4; ++tmi) {
        const __bf16* aHi = ldsHi + (tmi * 16 + lrow) * Dk;
        const __bf16* aLo = ldsLo + (tmi * 16 + lrow) * Dk;
        v16bf ah = frag16(aHi + o0, aHi + o1);
        v16bf al = frag16(aLo + o0, aLo + o1);
        v8f& c = (tmi == 0) ? c0 : (tmi == 1) ? c1 : (tmi == 2) ? c2 : c3;
        c = __builtin_amdgcn_wmma_f32_16x16x32_bf16(false, ah, false, bh,
                                                    (short)0, c, false, false);
        c = __builtin_amdgcn_wmma_f32_16x16x32_bf16(false, ah, false, bl,
                                                    (short)0, c, false, false);
        c = __builtin_amdgcn_wmma_f32_16x16x32_bf16(false, al, false, bh,
                                                    (short)0, c, false, false);
      }
    }
    const float e2n = e2[n];
#pragma unroll
    for (int tmi = 0; tmi < 4; ++tmi) {
      const v8f& c = (tmi == 0) ? c0 : (tmi == 1) ? c1 : (tmi == 2) ? c2 : c3;
#pragma unroll
      for (int r = 0; r < 8; ++r) {        // C layout: row = r + 8*half, col n
        float sc = e2n - 2.0f * c[r];      // x^2 omitted: constant per row
        if (sc < best[tmi][r] || (sc == best[tmi][r] && n < bidx[tmi][r])) {
          best[tmi][r] = sc; bidx[tmi][r] = n;
        }
      }
    }
  }

  // reduce argmin across the 16 lanes sharing each row, stash per-wave winner
#pragma unroll
  for (int tmi = 0; tmi < 4; ++tmi) {
#pragma unroll
    for (int r = 0; r < 8; ++r) {
      float bd = best[tmi][r]; int bi = bidx[tmi][r];
      for (int off = 1; off < 16; off <<= 1) {
        float od = __shfl_xor(bd, off, 32);
        int   oi = __shfl_xor(bi, off, 32);
        if (od < bd || (od == bd && oi < bi)) { bd = od; bi = oi; }
      }
      if (lrow == 0) {
        int m = tmi * 16 + half * 8 + r;   // row within WG tile
        redD[m * 4 + wave] = bd;
        redI[m * 4 + wave] = bi;
      }
    }
  }
  __syncthreads();

  // final 4-way reduction across waves (wave order == ascending code index)
  if (threadIdx.x < 64) {
    int m = threadIdx.x;
    float bd = redD[m * 4 + 0]; int bi = redI[m * 4 + 0];
#pragma unroll
    for (int w = 1; w < 4; ++w) {
      float od = redD[m * 4 + w]; int oi = redI[m * 4 + w];
      if (od < bd || (od == bd && oi < bi)) { bd = od; bi = oi; }
    }
    int row = rowBase + m;
    codes_i[row] = bi;
    codes_f[row] = (float)bi;
  }
}

// --- kernel 3: quantized = emb[code] scattered to (B, D, T) ----------------
__global__ __launch_bounds__(256)
void vq_gather(const int* __restrict__ codes,
               const float* __restrict__ es,
               const float* __restrict__ inv,
               float* __restrict__ out) {
  size_t i = (size_t)blockIdx.x * 256 + threadIdx.x;   // over B*D*T
  int t = (int)(i % Tk);
  size_t bd = i / Tk;
  int d = (int)(bd % Dk);
  int b = (int)(bd / Dk);
  int c = codes[b * Tk + t];
  out[i] = es[(size_t)c * Dk + d] * inv[c];
}

extern "C" void kernel_launch(void* const* d_in, const int* in_sizes, int n_in,
                              void* d_out, int out_size, void* d_ws, size_t ws_size,
                              hipStream_t stream) {
  (void)in_sizes; (void)n_in; (void)out_size; (void)ws_size;
  const float* x     = (const float*)d_in[0];   // (16, 512, 1024)
  const float* es    = (const float*)d_in[1];   // (8192, 512)
  const float* usage = (const float*)d_in[2];   // (8192,)

  char* ws = (char*)d_ws;
  __bf16* hi   = (__bf16*)(ws + OFF_HI);
  __bf16* lo   = (__bf16*)(ws + OFF_LO);
  float*  e2   = (float*)(ws + OFF_E2);
  float*  inv  = (float*)(ws + OFF_INV);
  int*    ci   = (int*)(ws + OFF_CODES);

  float* out     = (float*)d_out;
  float* codes_f = out;             // first 16384 = codes (B,1,T)
  float* quant   = out + Nrows;     // next 8388608 = quantized (B,D,T)

  hipFuncSetAttribute(reinterpret_cast<const void*>(vq_argmin),
                      hipFuncAttributeMaxDynamicSharedMemorySize, kLdsBytes);

  vq_prep_codebook<<<Kc, 128, 0, stream>>>(es, usage, hi, lo, e2, inv);
  vq_argmin<<<Nrows / 64, 128, kLdsBytes, stream>>>(x, hi, lo, e2, ci, codes_f);
  vq_gather<<<((size_t)Bk * Dk * Tk) / 256, 256, 0, stream>>>(ci, es, inv, quant);
}
